// MultiHeadAttention_11312943857760
// MI455X (gfx1250) — matrix operand order repaired and checked by reading, not compiled, not run on hardware
//
#include <hip/hip_runtime.h>
#include <stdint.h>

// ---------------------------------------------------------------- constants
#define Bsz   4
#define Lsz   1024
#define Dsz   1024
#define Hsz   16
#define DKsz  64
#define NTOK  (Bsz * Lsz)          // 4096 rows

typedef __attribute__((ext_vector_type(16))) __bf16 v16bf;
typedef __attribute__((ext_vector_type(8)))  float  v8f;

// fp32 -> bf16 round-to-nearest-even
__device__ __forceinline__ unsigned short f2bf(float f) {
  union { float f; unsigned int u; } x; x.f = f;
  unsigned int u = x.u;
  u += 0x7fffu + ((u >> 16) & 1u);
  return (unsigned short)(u >> 16);
}

__device__ __forceinline__ v8f vzero8() {
  v8f z;
#pragma unroll
  for (int i = 0; i < 8; ++i) z[i] = 0.0f;
  return z;
}

// Async memory->LDS copy of 8 bytes, GVS addressing:
// mem_addr = SGPR_base(u64) + VGPR_offset(u32).  LDS dest from VDST VGPR.
// ldsaddr: low 32 bits of generic LDS pointer == LDS byte offset (ISA 10.2).
__device__ __forceinline__ void async_copy_b64_s(void* lds, const void* sbase,
                                                 unsigned int voffBytes) {
  unsigned int ldsaddr = (unsigned int)(uintptr_t)lds;
  asm volatile("global_load_async_to_lds_b64 %0, %1, %2"
               :
               : "v"(ldsaddr), "v"(voffBytes), "s"(sbase)
               : "memory");
}

__device__ __forceinline__ void wait_async0() {
  asm volatile("s_wait_asynccnt 0x0" ::: "memory");
}

// A fragment: 16x32 bf16, LDS tile row-major [M][ks] (K contiguous).
// Element j of v16bf -> K = (j&7) + 8*hi + 16*(j>>3), M = lane&15.
__device__ __forceinline__ v16bf frag_a(const unsigned short* At, int lane, int ks) {
  const unsigned short* row = At + (lane & 15) * ks + ((lane >> 4) << 3);
  union { v16bf v; unsigned int u[8]; } r;
#pragma unroll
  for (int t = 0; t < 8; ++t) {
    int kk = ((t & 3) << 1) + ((t >> 2) << 4);
    r.u[t] = *(const unsigned int*)(row + kk);
  }
  return r.v;
}

// B fragment: 32x16 bf16 (KxN), LDS tile stored transposed [N][ks] (K contiguous).
// Element j -> K = (j&7) + 8*hi + 16*(j>>3), N = lane&15.
__device__ __forceinline__ v16bf frag_b(const unsigned short* Bt, int lane, int ks) {
  const unsigned short* col = Bt + (lane & 15) * ks + ((lane >> 4) << 3);
  union { v16bf v; unsigned int u[8]; } r;
#pragma unroll
  for (int t = 0; t < 8; ++t) r.u[t] = *(const unsigned int*)(col + ((t & 3) << 1) + ((t >> 2) << 4));
  return r.v;
}

__device__ __forceinline__ v8f wmma_bf16(v16bf a, v16bf b, v8f c) {
  // (neg_a, A, neg_b, B, c_mod, C, reuse_a, reuse_b)
  return __builtin_amdgcn_wmma_f32_16x16x32_bf16(false, a, false, b, (short)0, c,
                                                 false, false);
}

// ============================================================ K1 / K4 GEMM
// C[4096,1024] = A[4096,1024] @ B[1024,1024]
// EPI==0: C *= alpha, store bf16 (projections).  AT = float (convert on load).
// EPI==1: C += resid, LayerNorm(row), store fp32. AT = ushort (bf16 already).
template <int EPI, typename AT>
__global__ __launch_bounds__(256) void gemm16row(
    const AT* __restrict__ A, const float* __restrict__ B0,
    const float* __restrict__ B1, const int* __restrict__ flagp, float alpha,
    const float* __restrict__ resid, const float* __restrict__ lng,
    const float* __restrict__ lnb, void* __restrict__ OutP) {
  constexpr int N = 1024, K = 1024, KS = 32, KP = 40, NCH = 256;
  const float* __restrict__ Bw = (flagp[0] != 0) ? B1 : B0;

  __shared__ __align__(16) unsigned short At[16 * KP];
  __shared__ __align__(16) unsigned short Bt[NCH * KP];
  __shared__ float redS[8 * 16];
  __shared__ float redQ[8 * 16];
  __shared__ float rowA[16];
  __shared__ float rowB[16];

  const int tid = threadIdx.x;
  const int lane = tid & 31, w = tid >> 5, hi = lane >> 4, l15 = lane & 15;
  const int row0 = blockIdx.x * 16;

  v8f acc[8];
#pragma unroll
  for (int i = 0; i < 8; ++i) acc[i] = vzero8();

#pragma unroll
  for (int nc = 0; nc < 4; ++nc) {
    const int n0 = nc * NCH;
    for (int k0 = 0; k0 < K; k0 += KS) {
      // stage A tile 16x32
#pragma unroll
      for (int i = 0; i < 2; ++i) {
        int e = i * 256 + tid;
        int m = e >> 5, kk = e & 31;
        if constexpr (sizeof(AT) == 4)
          At[m * KP + kk] = f2bf(((const float*)A)[(size_t)(row0 + m) * K + k0 + kk]);
        else
          At[m * KP + kk] = ((const unsigned short*)A)[(size_t)(row0 + m) * K + k0 + kk];
      }
      // stage B tile 32x256 transposed into [n][k]; float2 loads along N
#pragma unroll
      for (int i = 0; i < 16; ++i) {
        int e = i * 256 + tid;        // 4096 float2 groups
        int kk = e >> 7, n = (e & 127) * 2;
        float2 f = *(const float2*)&Bw[(size_t)(k0 + kk) * N + n0 + n];
        Bt[n * KP + kk] = f2bf(f.x);
        Bt[(n + 1) * KP + kk] = f2bf(f.y);
      }
      // prefetch next K-tile of B into cache (global_prefetch_b8)
      if (k0 + KS < K)
        __builtin_prefetch(&Bw[(size_t)(k0 + KS + (tid & 31)) * N + n0 + (tid >> 5) * 32],
                           0, 0);
      __syncthreads();
      v16bf af = frag_a(At, lane, KP);
#pragma unroll
      for (int t = 0; t < 2; ++t) {
        v16bf bf = frag_b(Bt + (w * 32 + t * 16) * KP, lane, KP);
        acc[nc * 2 + t] = wmma_bf16(af, bf, acc[nc * 2 + t]);
      }
      __syncthreads();
    }
  }

  if constexpr (EPI == 0) {
    unsigned short* Out = (unsigned short*)OutP;
#pragma unroll
    for (int nc = 0; nc < 4; ++nc)
#pragma unroll
      for (int t = 0; t < 2; ++t) {
        int col = nc * NCH + w * 32 + t * 16 + l15;
#pragma unroll
        for (int r = 0; r < 8; ++r) {
          int grow = row0 + r + hi * 8;
          Out[(size_t)grow * N + col] = f2bf(acc[nc * 2 + t][r] * alpha);
        }
      }
  } else {
    float* Out = (float*)OutP;
    float ps[8], pq[8];
#pragma unroll
    for (int r = 0; r < 8; ++r) { ps[r] = 0.0f; pq[r] = 0.0f; }
#pragma unroll
    for (int nc = 0; nc < 4; ++nc)
#pragma unroll
      for (int t = 0; t < 2; ++t) {
        int col = nc * NCH + w * 32 + t * 16 + l15;
#pragma unroll
        for (int r = 0; r < 8; ++r) {
          int grow = row0 + r + hi * 8;
          float val = acc[nc * 2 + t][r] + resid[(size_t)grow * N + col];
          acc[nc * 2 + t][r] = val;
          ps[r] += val;
          pq[r] += val * val;
        }
      }
#pragma unroll
    for (int r = 0; r < 8; ++r)
#pragma unroll
      for (int off = 1; off < 16; off <<= 1) {
        ps[r] += __shfl_xor(ps[r], off, 16);
        pq[r] += __shfl_xor(pq[r], off, 16);
      }
    if (l15 == 0) {
#pragma unroll
      for (int r = 0; r < 8; ++r) {
        redS[w * 16 + hi * 8 + r] = ps[r];
        redQ[w * 16 + hi * 8 + r] = pq[r];
      }
    }
    __syncthreads();
    if (tid < 16) {
      float s = 0.0f, q2 = 0.0f;
#pragma unroll
      for (int j = 0; j < 8; ++j) { s += redS[j * 16 + tid]; q2 += redQ[j * 16 + tid]; }
      rowA[tid] = s;
      rowB[tid] = q2;
    }
    __syncthreads();
    float mu[8], rs[8];
#pragma unroll
    for (int r = 0; r < 8; ++r) {
      float m = rowA[hi * 8 + r] * (1.0f / N);
      float var = rowB[hi * 8 + r] * (1.0f / N) - m * m;
      mu[r] = m;
      rs[r] = rsqrtf(var + 1e-6f);
    }
#pragma unroll
    for (int nc = 0; nc < 4; ++nc)
#pragma unroll
      for (int t = 0; t < 2; ++t) {
        int col = nc * NCH + w * 32 + t * 16 + l15;
        float gv = lng[col], bv = lnb[col];
#pragma unroll
        for (int r = 0; r < 8; ++r) {
          int grow = row0 + r + hi * 8;
          Out[(size_t)grow * N + col] =
              (acc[nc * 2 + t][r] - mu[r]) * rs[r] * gv + bv;
        }
      }
  }
}

// ============================================================ K2: QK^T + softmax
// Per (b,h): 16-row stripe x all 1024 cols, fused softmax, fp32 probs -> d_out.
// A/B tiles are bf16->bf16 copies: staged with async global->LDS (ASYNCcnt),
// GVS addressing (uniform SGPR base + 32-bit lane offset).
__global__ __launch_bounds__(256) void attn_kernel(
    const unsigned short* __restrict__ qh, const unsigned short* __restrict__ kh,
    float* __restrict__ attn) {
  constexpr int KP = 72;  // 64 + pad (144B rows, 8B aligned)
  __shared__ __align__(16) unsigned short At[16 * KP];
  __shared__ __align__(16) unsigned short Bt[256 * KP];
  __shared__ float red[8 * 16];
  __shared__ float rowr[16];

  const int tid = threadIdx.x, lane = tid & 31, w = tid >> 5;
  const int hi = lane >> 4, l15 = lane & 15;
  const int row0 = blockIdx.x * 16;
  const int bh = blockIdx.y;
  const int b = bh >> 4, h = bh & 15;
  const size_t tokBase = (size_t)b * Lsz;
  const int headOff = h * DKsz;

  // block-uniform bases (land in SGPR pairs for GVS async loads)
  const unsigned short* qbase = qh + (tokBase + row0) * 1024 + headOff;
  const unsigned short* kbase = kh + tokBase * 1024 + headOff;

  // stage A (16x64 bf16) asynchronously: 256 groups of 4 elements
  {
    int m = tid >> 4, k4 = (tid & 15) * 4;
    async_copy_b64_s(At + m * KP + k4, qbase, (unsigned)(m * 2048 + k4 * 2));
  }

  v8f acc[8];
#pragma unroll
  for (int i = 0; i < 8; ++i) acc[i] = vzero8();

  v16bf af0, af1;
#pragma unroll
  for (int nc = 0; nc < 4; ++nc) {
    const int n0 = nc * 256;
    __syncthreads();  // previous chunk's readers are done with Bt
    // stage B: kh rows n0..n0+255 (natural [N][K] layout), async to LDS
#pragma unroll
    for (int i = 0; i < 16; ++i) {
      int e = i * 256 + tid;  // 4096 groups of 4 = 16384 elements
      int n = e >> 4, k4 = (e & 15) * 4;
      async_copy_b64_s(Bt + n * KP + k4, kbase,
                       (unsigned)((n0 + n) * 2048 + k4 * 2));
    }
    wait_async0();
    __syncthreads();
    if (nc == 0) { af0 = frag_a(At, lane, KP); af1 = frag_a(At + 32, lane, KP); }
#pragma unroll
    for (int t = 0; t < 2; ++t) {
      const unsigned short* bp = Bt + (w * 32 + t * 16) * KP;
      v16bf bf0 = frag_b(bp, lane, KP);
      v16bf bf1 = frag_b(bp + 32, lane, KP);
      acc[nc * 2 + t] = wmma_bf16(af1, bf1, wmma_bf16(af0, bf0, acc[nc * 2 + t]));
    }
  }

  // ---- row softmax (scale already folded into qh) ----
  float pm[8];
#pragma unroll
  for (int r = 0; r < 8; ++r) pm[r] = -3.4e38f;
#pragma unroll
  for (int i = 0; i < 8; ++i)
#pragma unroll
    for (int r = 0; r < 8; ++r) pm[r] = fmaxf(pm[r], acc[i][r]);
#pragma unroll
  for (int r = 0; r < 8; ++r)
#pragma unroll
    for (int off = 1; off < 16; off <<= 1)
      pm[r] = fmaxf(pm[r], __shfl_xor(pm[r], off, 16));
  if (l15 == 0) {
#pragma unroll
    for (int r = 0; r < 8; ++r) red[w * 16 + hi * 8 + r] = pm[r];
  }
  __syncthreads();
  if (tid < 16) {
    float m = -3.4e38f;
#pragma unroll
    for (int j = 0; j < 8; ++j) m = fmaxf(m, red[j * 16 + tid]);
    rowr[tid] = m;
  }
  __syncthreads();
  float rm[8];
#pragma unroll
  for (int r = 0; r < 8; ++r) rm[r] = rowr[hi * 8 + r];

  float psum[8];
#pragma unroll
  for (int r = 0; r < 8; ++r) psum[r] = 0.0f;
#pragma unroll
  for (int i = 0; i < 8; ++i)
#pragma unroll
    for (int r = 0; r < 8; ++r) {
      float e = __expf(acc[i][r] - rm[r]);
      acc[i][r] = e;
      psum[r] += e;
    }
#pragma unroll
  for (int r = 0; r < 8; ++r)
#pragma unroll
    for (int off = 1; off < 16; off <<= 1) psum[r] += __shfl_xor(psum[r], off, 16);
  __syncthreads();  // all rm reads done before red reuse
  if (l15 == 0) {
#pragma unroll
    for (int r = 0; r < 8; ++r) red[w * 16 + hi * 8 + r] = psum[r];
  }
  __syncthreads();
  if (tid < 16) {
    float s = 0.0f;
#pragma unroll
    for (int j = 0; j < 8; ++j) s += red[j * 16 + tid];
    rowr[tid] = s;
  }
  __syncthreads();
  float inv[8];
#pragma unroll
  for (int r = 0; r < 8; ++r) inv[r] = 1.0f / rowr[hi * 8 + r];

  const size_t base = (size_t)bh * Lsz * Lsz;
#pragma unroll
  for (int nc = 0; nc < 4; ++nc)
#pragma unroll
    for (int t = 0; t < 2; ++t) {
      int col = nc * 256 + w * 32 + t * 16 + l15;
#pragma unroll
      for (int r = 0; r < 8; ++r) {
        int grow = row0 + r + hi * 8;
        attn[base + (size_t)grow * Lsz + col] = acc[nc * 2 + t][r] * inv[r];
      }
    }
}

// ============================================================ K3: P @ V
// Per (b,h): 64-row block x 64 cols, K=1024; P fp32 from d_out -> bf16 tiles.
__global__ __launch_bounds__(128) void pv_kernel(
    const float* __restrict__ attn, const unsigned short* __restrict__ vh,
    unsigned short* __restrict__ ao) {
  constexpr int KP = 40;
  __shared__ __align__(16) unsigned short At[64 * KP];
  __shared__ __align__(16) unsigned short Bt[64 * KP];

  const int tid = threadIdx.x, lane = tid & 31, w = tid >> 5;
  const int hi = lane >> 4, l15 = lane & 15;
  const int row0 = blockIdx.x * 64;
  const int bh = blockIdx.y, b = bh >> 4, h = bh & 15;
  const size_t abase = (size_t)bh * Lsz * Lsz;

  v8f acc[4];
#pragma unroll
  for (int i = 0; i < 4; ++i) acc[i] = vzero8();

  for (int k0 = 0; k0 < Lsz; k0 += 32) {
    __syncthreads();
    // stage A: 64x32 fp32 probs -> bf16
#pragma unroll
    for (int i = 0; i < 16; ++i) {
      int e = i * 128 + tid;  // 2048 elements
      int m = e >> 5, kk = e & 31;
      At[m * KP + kk] = f2bf(attn[abase + (size_t)(row0 + m) * Lsz + k0 + kk]);
    }
    // stage B: vh[k0+kk][n] -> Bt[n][kk] (transpose; VGPR path)
#pragma unroll
    for (int i = 0; i < 16; ++i) {
      int e = i * 128 + tid;
      int kk = e >> 6, n = e & 63;
      Bt[n * KP + kk] = vh[((size_t)b * Lsz + k0 + kk) * 1024 + h * DKsz + n];
    }
    __syncthreads();
    v16bf af = frag_a(At + (w * 16) * KP, lane, KP);
#pragma unroll
    for (int t = 0; t < 4; ++t) {
      v16bf bf = frag_b(Bt + t * 16 * KP, lane, KP);
      acc[t] = wmma_bf16(af, bf, acc[t]);
    }
  }
  // store bf16 token-major: ao[(b*L + row)*1024 + h*64 + col]
#pragma unroll
  for (int t = 0; t < 4; ++t) {
    int col = t * 16 + l15;
#pragma unroll
    for (int r = 0; r < 8; ++r) {
      int grow = row0 + w * 16 + r + hi * 8;
      ao[((size_t)b * Lsz + grow) * 1024 + h * DKsz + col] = f2bf(acc[t][r]);
    }
  }
}

// ============================================================ host launcher
extern "C" void kernel_launch(void* const* d_in, const int* in_sizes, int n_in,
                              void* d_out, int out_size, void* d_ws, size_t ws_size,
                              hipStream_t stream) {
  const float* q  = (const float*)d_in[0];
  const float* k  = (const float*)d_in[1];
  const float* v  = (const float*)d_in[2];
  const float* Wt = (const float*)d_in[3];
  const float* Wi = (const float*)d_in[4];
  const float* Wc = (const float*)d_in[5];
  const float* Wf = (const float*)d_in[6];
  const float* lg = (const float*)d_in[7];
  const float* lb = (const float*)d_in[8];
  const int* flag = (const int*)d_in[9];

  float* outF  = (float*)d_out;                       // [4096,1024]
  float* attnF = outF + (size_t)NTOK * Dsz;           // [64,1024,1024]

  const size_t MAT = (size_t)NTOK * Dsz;              // 4M bf16 elems = 8MB
  unsigned short* qh = (unsigned short*)d_ws;
  unsigned short* kh = qh + MAT;
  unsigned short* vh = kh + MAT;
  unsigned short* ao = vh + MAT;                      // total ws use: 32 MB

  const float qscale = 0.125f;  // 1/sqrt(64), folded into q projection

  // projections (flag selects: q -> W_img : W_token, k/v -> W_cls : W_img)
  gemm16row<0, float><<<NTOK / 16, 256, 0, stream>>>(q, Wt, Wi, flag, qscale,
                                                     nullptr, nullptr, nullptr, qh);
  gemm16row<0, float><<<NTOK / 16, 256, 0, stream>>>(k, Wi, Wc, flag, 1.0f,
                                                     nullptr, nullptr, nullptr, kh);
  gemm16row<0, float><<<NTOK / 16, 256, 0, stream>>>(v, Wi, Wc, flag, 1.0f,
                                                     nullptr, nullptr, nullptr, vh);
  // attention scores + softmax -> d_out attn region
  attn_kernel<<<dim3(Lsz / 16, Bsz * Hsz), 256, 0, stream>>>(qh, kh, attnF);
  // P @ V -> ws (bf16, token-major)
  pv_kernel<<<dim3(Lsz / 64, Bsz * Hsz), 128, 0, stream>>>(attnF, vh, ao);
  // fc + residual + LayerNorm -> d_out out region
  gemm16row<1, unsigned short><<<NTOK / 16, 256, 0, stream>>>(
      ao, Wf, Wf, flag, 1.0f, q, lg, lb, outF);
}
